// ROCLoss_42133629174003
// MI455X (gfx1250) — compile-verified, matrix operand order checked by reading, not run on hardware
//
#include <hip/hip_runtime.h>
#include <math.h>

// Problem constants (fixed by the reference): B=8192 examples, K=512 timesteps,
// C=16 candidates. Output = scalar mean NLL.
#define B_N 8192
#define K_N 512
#define C_N 16
#define ROWS (B_N * C_N)            // 131072 (i,j) rows
#define ROW_BYTES (K_N * 4)         // 2048 bytes of mask per row
#define WAVES_PER_BLOCK 8           // 256 threads, wave32
#define ROWS_PER_WAVE 16            // one WMMA M-dimension worth of rows
#define ROWS_PER_BLOCK (WAVES_PER_BLOCK * ROWS_PER_WAVE)  // 128
#define BLOCKS1 (ROWS / ROWS_PER_BLOCK)                   // 1024
#define BLOCKS2 (B_N / 256)                               // 32

typedef __attribute__((ext_vector_type(8))) int v8i;

// ---------------------------------------------------------------------------
// Kernel 1: per (i,j) row, count leading ones of mask via WMMA byte-sum
// (0/1 int32 -> byte sum == int sum), then gather logits[i, cnt-1+?, j].
// counter = sum - 1 (mask rows are contiguous leading-one runs, length>=1).
// Each wave owns 16 rows; lane L (L<16) and lane L+16 jointly feed 64 bytes
// of row M=L per WMMA step (B = all-ones, so K-permutation within a row is
// irrelevant to the sum). 32 accumulating v_wmma_i32_16x16x64_iu8 per wave.
// ---------------------------------------------------------------------------
__global__ __launch_bounds__(256) void
wmma_count_gather_kernel(const int* __restrict__ mask,
                         const float* __restrict__ logits,
                         float* __restrict__ adjusted) {
  const int tid  = threadIdx.x;
  const int lane = tid & 31;
  const int wave = tid >> 5;
  const int rowBase = blockIdx.x * ROWS_PER_BLOCK + wave * ROWS_PER_WAVE;
  const int m    = lane & 15;   // WMMA A-matrix row this lane feeds
  const int half = lane >> 4;   // which 32-byte half of the 64-byte chunk

  // byte pointer into this lane's slice of its row's mask data
  const char* base = (const char*)mask
                   + (size_t)(rowBase + m) * ROW_BYTES
                   + (size_t)half * 32;

  const int kOnes = 0x01010101;
  v8i ones = {kOnes, kOnes, kOnes, kOnes, kOnes, kOnes, kOnes, kOnes};
  v8i acc  = {0, 0, 0, 0, 0, 0, 0, 0};

  // 2048 bytes per row / 64 bytes per WMMA = 32 steps. Each step: one 32-byte
  // (2 x b128) load per lane + one accumulating IU8 WMMA.
#pragma unroll 4
  for (int step = 0; step < ROW_BYTES / 64; ++step) {
    v8i a = *(const v8i*)(base + (size_t)step * 64);  // 32B aligned
    acc = __builtin_amdgcn_wmma_i32_16x16x64_iu8(
        /*sgn_a=*/false, a, /*sgn_b=*/false, ones, acc,
        /*reuse_a=*/false, /*reuse_b=*/false);
  }

  // i32 16x16 C/D layout: lanes 0-15 VGPR v -> M=v (N=lane);
  // lanes 16-31 VGPR v -> M=8+v (N=lane-16).
  // Lane 0 holds rows rowBase+0..7, lane 16 holds rows rowBase+8..15.
  if (m == 0) {
    const int mb = half * 8;
#pragma unroll
    for (int v = 0; v < 8; ++v) {
      int r   = rowBase + mb + v;
      int cnt = acc[v] - 1;                    // leading-ones count - 1
      int i   = r >> 4;                        // example index
      int j   = r & 15;                        // candidate index
      adjusted[r] = logits[((size_t)i * K_N + (size_t)cnt) * C_N + j];
    }
  }
}

// ---------------------------------------------------------------------------
// Kernel 2: per example, log-softmax over 16 candidates, NLL of target,
// fixed-order block reduction -> 32 partial sums.
// ---------------------------------------------------------------------------
__global__ __launch_bounds__(256) void
nll_partial_kernel(const float* __restrict__ adjusted,
                   const int* __restrict__ target,
                   float* __restrict__ partials) {
  __shared__ float red[256];
  const int i = blockIdx.x * 256 + threadIdx.x;

  float a[16];
  const float4* p = (const float4*)(adjusted + (size_t)i * 16);
#pragma unroll
  for (int q = 0; q < 4; ++q) {
    float4 v4 = p[q];
    a[4 * q + 0] = v4.x;
    a[4 * q + 1] = v4.y;
    a[4 * q + 2] = v4.z;
    a[4 * q + 3] = v4.w;
  }

  float mx = a[0];
#pragma unroll
  for (int q = 1; q < 16; ++q) mx = fmaxf(mx, a[q]);
  float s = 0.f;
#pragma unroll
  for (int q = 0; q < 16; ++q) s += expf(a[q] - mx);
  const float lse = mx + logf(s);

  const int t = target[i];
  const float at = adjusted[(size_t)i * 16 + (size_t)t];  // avoid scratch index
  red[threadIdx.x] = lse - at;
  __syncthreads();

  // deterministic tree reduction
  for (int off = 128; off > 0; off >>= 1) {
    if (threadIdx.x < off) red[threadIdx.x] += red[threadIdx.x + off];
    __syncthreads();
  }
  if (threadIdx.x == 0) partials[blockIdx.x] = red[0];
}

// ---------------------------------------------------------------------------
// Kernel 3: fixed-order final sum of 32 partials, divide by B. Deterministic.
// ---------------------------------------------------------------------------
__global__ void final_kernel(const float* __restrict__ partials,
                             float* __restrict__ out) {
  if (threadIdx.x == 0) {
    float s = 0.f;
    for (int q = 0; q < BLOCKS2; ++q) s += partials[q];
    out[0] = s / (float)B_N;
  }
}

extern "C" void kernel_launch(void* const* d_in, const int* in_sizes, int n_in,
                              void* d_out, int out_size, void* d_ws, size_t ws_size,
                              hipStream_t stream) {
  (void)in_sizes; (void)n_in; (void)out_size; (void)ws_size;

  const float* logits = (const float*)d_in[0];  // [B, K, C] f32
  const int*   target = (const int*)d_in[1];    // [B] i32
  const int*   mask   = (const int*)d_in[2];    // [B, C, K] i32 (0/1)

  float* adjusted = (float*)d_ws;               // [B*C] f32 = 512 KB
  float* partials = (float*)d_ws + ROWS;        // [32] f32
  float* out      = (float*)d_out;              // scalar

  wmma_count_gather_kernel<<<BLOCKS1, 256, 0, stream>>>(mask, logits, adjusted);
  nll_partial_kernel<<<BLOCKS2, 256, 0, stream>>>(adjusted, target, partials);
  final_kernel<<<1, 32, 0, stream>>>(partials, out);
}
